// MultiHeadAttention_24326694765180
// MI455X (gfx1250) — matrix operand order repaired and checked by reading, not compiled, not run on hardware
//
#include <hip/hip_runtime.h>
#include <hip/hip_bf16.h>

// ---------------------------------------------------------------------------
// MI455X (gfx1250): wave32, v_wmma_f32_16x16x32_f16 GEMMs fed by the Tensor
// Data Mover (tensor_load_to_lds, TENSORcnt double-buffered pipeline).
// Problem: S=8192, D=1024, CLASSES=7.  Outputs: o[S,D] f32, c[S,7] f32,
// attention[S,S] f32.  HBM-bound on the 268MB attention materialization.
// ---------------------------------------------------------------------------

typedef _Float16 half8 __attribute__((ext_vector_type(8)));
typedef _Float16 v16h  __attribute__((ext_vector_type(16)));
typedef float    v8f   __attribute__((ext_vector_type(8)));
typedef unsigned u32x4 __attribute__((ext_vector_type(4)));
typedef unsigned u32x8 __attribute__((ext_vector_type(8)));

#define SEQ_LEN  8192
#define DIM      1024
#define NCLASSES 7

#define TILE_M 128
#define TILE_N 64
#define TILE_K 32

// ---------------------------------------------------------------------------
// LDS byte offset of a __shared__ object (generic -> addrspace(3) cast gives
// the 32-bit LDS offset the TDM descriptor wants).
// ---------------------------------------------------------------------------
__device__ __forceinline__ unsigned lds_addr_of(const void* p) {
    return (unsigned)(size_t)(__attribute__((address_space(3))) const void*)p;
}

// ---------------------------------------------------------------------------
// TDM 2-D tile load: issue TENSOR_LOAD_TO_LDS with a 2-group D#
// (CDNA5 ISA).  data_size = 2 bytes (f16).
//   tile = tile_d1 rows x tile_d0 elements, row stride = stride_elems,
//   written contiguously (row-major) into LDS at lds_addr.
// ---------------------------------------------------------------------------
__device__ __forceinline__ void tdm_load_2d(unsigned lds_addr, const void* gaddr,
                                            unsigned tensor_d0, unsigned tensor_d1,
                                            unsigned tile_d0, unsigned tile_d1,
                                            unsigned stride_elems)
{
    unsigned long long ga = (unsigned long long)gaddr;
    u32x4 g0;
    g0[0] = 1u;                                          // count=1, user D#
    g0[1] = lds_addr;                                    // LDS byte address
    g0[2] = (unsigned)(ga & 0xFFFFFFFFu);                // global_addr[31:0]
    g0[3] = (unsigned)((ga >> 32) & 0x01FFFFFFu)         // global_addr[56:32]
          | (2u << 30);                                  // type = 2 (image)
    u32x8 g1;
    g1[0] = (1u << 16);                                  // data_size = 2B
    g1[1] = (tensor_d0 & 0xFFFFu) << 16;                 // tensor_dim0[15:0]
    g1[2] = ((tensor_d0 >> 16) & 0xFFFFu)                // tensor_dim0[31:16]
          | ((tensor_d1 & 0xFFFFu) << 16);               // tensor_dim1[15:0]
    g1[3] = ((tensor_d1 >> 16) & 0xFFFFu)                // tensor_dim1[31:16]
          | ((tile_d0 & 0xFFFFu) << 16);                 // tile_dim0
    g1[4] = (tile_d1 & 0xFFFFu);                         // tile_dim1 (dim2=0)
    g1[5] = stride_elems;                                // dim0_stride[31:0]
    g1[6] = 0u;                                          // stride hi / dim1_stride
    g1[7] = 0u;
    asm volatile("tensor_load_to_lds %0, %1" :: "s"(g0), "s"(g1) : "memory");
}

// ---------------------------------------------------------------------------
// Cast f32 -> f16 (grid stride)
// ---------------------------------------------------------------------------
__global__ void __launch_bounds__(256) cast_f16_kernel(const float* __restrict__ src,
                                                       _Float16* __restrict__ dst,
                                                       long long n) {
    long long i = (long long)blockIdx.x * blockDim.x + threadIdx.x;
    long long stride = (long long)gridDim.x * blockDim.x;
    for (; i < n; i += stride) dst[i] = (_Float16)src[i];
}

// Cast + transpose: W[K,N] f32 -> Wt[N,K] f16
__global__ void __launch_bounds__(256) cast_transpose_kernel(const float* __restrict__ W,
                                                             _Float16* __restrict__ Wt,
                                                             int K, int N) {
    long long idx = (long long)blockIdx.x * blockDim.x + threadIdx.x;
    long long total = (long long)K * N;
    if (idx >= total) return;
    int k = (int)(idx / N);
    int n = (int)(idx % N);
    Wt[(long long)n * K + k] = (_Float16)W[idx];
}

// ---------------------------------------------------------------------------
// Classifier head: c = x @ Wc + bc   (N=7 -> one wave per class, per row)
// ---------------------------------------------------------------------------
__global__ void __launch_bounds__(256) classifier_kernel(const float* __restrict__ x,
                                                         const float* __restrict__ Wc,
                                                         const float* __restrict__ bc,
                                                         float* __restrict__ out,
                                                         int d) {
    const int row  = blockIdx.x;
    const int lane = threadIdx.x & 31;
    const int cls  = threadIdx.x >> 5;
    if (cls >= NCLASSES) return;
    const float* xr = x + (long long)row * d;
    float s = 0.f;
    for (int i = lane; i < d; i += 32) s += xr[i] * Wc[(long long)i * NCLASSES + cls];
    #pragma unroll
    for (int off = 16; off > 0; off >>= 1) s += __shfl_down(s, off, 32);
    if (lane == 0) out[(long long)row * NCLASSES + cls] = s + bc[cls];
}

// ---------------------------------------------------------------------------
// WMMA helpers (layouts per CDNA5 ISA §7.12.2, wave32)
// ---------------------------------------------------------------------------
__device__ __forceinline__ v16h cat16(half8 lo, half8 hi) {
    return __builtin_shufflevector(lo, hi, 0, 1, 2, 3, 4, 5, 6, 7,
                                           8, 9, 10, 11, 12, 13, 14, 15);
}

__device__ __forceinline__ v8f wmma_f16(v16h a, v16h b, v8f c) {
    return __builtin_amdgcn_wmma_f32_16x16x32_f16(
        false, a, false, b, (short)0, c, false, false);
}

// ---------------------------------------------------------------------------
// TDM-staged WMMA GEMM:  C[M,N] = scale * (A[M,K] @ Bt[N,K]^T) + bias
// A row-major f16 (lda), Bt row-major f16 (ldb).  Block = 256 thr = 8 waves;
// block tile 128x64: wave w owns rows w*16..w*16+15, 4 WMMAs across 64 cols.
// Per K-step (32): wave 0 DMAs A-tile (128x32) + Bt-tile (64x32) into LDS
// via TENSOR_LOAD_TO_LDS, double buffered on TENSORcnt; all waves read
// fragments with ds_load_b128.
// ---------------------------------------------------------------------------
__global__ void __launch_bounds__(256) gemm_wmma_tdm_kernel(
    const _Float16* __restrict__ A, const _Float16* __restrict__ Bt,
    float* __restrict__ dstF, _Float16* __restrict__ dstH,
    const float* __restrict__ bias, float scale,
    int M, int N, int K, int lda, int ldb, int ldc)
{
    __shared__ alignas(16) _Float16 smA[2][TILE_M * TILE_K];   // 2 x 8KB
    __shared__ alignas(16) _Float16 smB[2][TILE_N * TILE_K];   // 2 x 4KB

    const int lane   = threadIdx.x & 31;
    const int wave   = threadIdx.x >> 5;
    const int rowBlk = blockIdx.x * TILE_M;
    const int colBlk = blockIdx.y * TILE_N;
    if (rowBlk >= M || colBlk >= N) return;     // block-uniform

    const unsigned ldsA[2] = { lds_addr_of(&smA[0][0]), lds_addr_of(&smA[1][0]) };
    const unsigned ldsB[2] = { lds_addr_of(&smB[0][0]), lds_addr_of(&smB[1][0]) };
    const _Float16* gA = A  + (size_t)rowBlk * lda;
    const _Float16* gB = Bt + (size_t)colBlk * ldb;

    // A fragment: lane holds row (lane&15); halves 0..7 = K kb+(lane>>4)*8+0..7,
    // halves 8..15 = those +16.  B fragment: lane holds col; same K layout
    // as A.
    const int lane15 = lane & 15;
    const int aK = (lane >> 4) << 3;
    const int bK = (lane >> 4) << 3;
    const int aRowOff = (wave * 16 + lane15) * TILE_K + aK;

    v8f acc[4] = {v8f{}, v8f{}, v8f{}, v8f{}};

    const int steps = K / TILE_K;
    if (wave == 0) {     // prologue DMA: tile 0 -> buffer 0
        tdm_load_2d(ldsA[0], gA, K, M, TILE_K, TILE_M, lda);
        tdm_load_2d(ldsB[0], gB, K, N, TILE_K, TILE_N, ldb);
    }

    for (int i = 0; i < steps; ++i) {
        const int cur = i & 1;
        if (wave == 0) {
            if (i + 1 < steps) {    // DMA next tile into other buffer
                const int kb = (i + 1) * TILE_K;
                tdm_load_2d(ldsA[cur ^ 1], gA + kb, K, M, TILE_K, TILE_M, lda);
                tdm_load_2d(ldsB[cur ^ 1], gB + kb, K, N, TILE_K, TILE_N, ldb);
                __builtin_amdgcn_s_wait_tensorcnt(2);  // pair i complete
            } else {
                __builtin_amdgcn_s_wait_tensorcnt(0);
            }
        }
        __syncthreads();            // tile i visible to all waves

        half8 alo = *(const half8*)&smA[cur][aRowOff];
        half8 ahi = *(const half8*)&smA[cur][aRowOff + 16];
        v16h a = cat16(alo, ahi);
        #pragma unroll
        for (int j = 0; j < 4; ++j) {
            const _Float16* pb = &smB[cur][(j * 16 + lane15) * TILE_K + bK];
            v16h b = cat16(*(const half8*)pb, *(const half8*)(pb + 16));
            acc[j] = wmma_f16(a, b, acc[j]);    // A fragment reused x4
        }
        __syncthreads();            // tile i consumed; buffer reusable
    }

    // D layout: lanes 0-15 -> M=rbase+i, N=col; lanes 16-31 -> M=rbase+i (their rbase includes +8).
    const int rbase = rowBlk + wave * 16 + ((lane >> 4) << 3);
    #pragma unroll
    for (int j = 0; j < 4; ++j) {
        const int col = colBlk + j * 16 + lane15;
        const float bval = bias ? bias[col] : 0.f;
        #pragma unroll
        for (int i = 0; i < 8; ++i) {
            const float v = acc[j][i] * scale + bval;
            if (dstF) dstF[(size_t)(rbase + i) * ldc + col] = v;
            if (dstH) dstH[(size_t)(rbase + i) * ldc + col] = (_Float16)v;
        }
    }
}

// ---------------------------------------------------------------------------
// LayerNorm over d=1024, reading f32 GEMM result, writing f16 (optionally
// transposed so the result is directly usable as a Bt GEMM operand).
// ---------------------------------------------------------------------------
__global__ void __launch_bounds__(256) ln_cast_kernel(
    const float* __restrict__ src, const float* __restrict__ g,
    const float* __restrict__ beta, _Float16* __restrict__ dst,
    int d, int transposed, int ldt)
{
    __shared__ float buf[256];
    const int row = blockIdx.x;
    const int tid = threadIdx.x;
    const float* p = src + (long long)row * d;

    float s = 0.f;
    for (int i = tid; i < d; i += 256) s += p[i];
    buf[tid] = s; __syncthreads();
    for (int st = 128; st > 0; st >>= 1) {
        if (tid < st) buf[tid] += buf[tid + st];
        __syncthreads();
    }
    const float mean = buf[0] / (float)d;
    __syncthreads();

    float v = 0.f;
    for (int i = tid; i < d; i += 256) { float t = p[i] - mean; v += t * t; }
    buf[tid] = v; __syncthreads();
    for (int st = 128; st > 0; st >>= 1) {
        if (tid < st) buf[tid] += buf[tid + st];
        __syncthreads();
    }
    const float rstd = rsqrtf(buf[0] / (float)d + 1e-5f);

    for (int i = tid; i < d; i += 256) {
        float y = (p[i] - mean) * rstd * g[i] + beta[i];
        if (transposed) dst[(long long)i * ldt + row] = (_Float16)y;
        else            dst[(long long)row * d + i]   = (_Float16)y;
    }
}

// ---------------------------------------------------------------------------
// Row softmax over S=8192: f32 logits in-place in the attention output region,
// plus an f16 transposed copy for the values GEMM.  Row cached in 32KB LDS.
// ---------------------------------------------------------------------------
__global__ void __launch_bounds__(256) softmax_kernel(float* __restrict__ attn,
                                                      _Float16* __restrict__ attnT,
                                                      int S)
{
    __shared__ float rowbuf[SEQ_LEN];
    __shared__ float red[256];
    const int row = blockIdx.x;
    const int tid = threadIdx.x;
    float* p = attn + (long long)row * S;

    float mx = -3.4e38f;
    for (int i = tid; i < S; i += 256) {
        float v = p[i];
        rowbuf[i] = v;
        mx = fmaxf(mx, v);
    }
    red[tid] = mx; __syncthreads();
    for (int st = 128; st > 0; st >>= 1) {
        if (tid < st) red[tid] = fmaxf(red[tid], red[tid + st]);
        __syncthreads();
    }
    mx = red[0]; __syncthreads();

    float sum = 0.f;
    for (int i = tid; i < S; i += 256) {
        float e = __expf(rowbuf[i] - mx);
        rowbuf[i] = e;
        sum += e;
    }
    red[tid] = sum; __syncthreads();
    for (int st = 128; st > 0; st >>= 1) {
        if (tid < st) red[tid] += red[tid + st];
        __syncthreads();
    }
    const float inv = 1.f / red[0];

    for (int i = tid; i < S; i += 256) {
        float a = rowbuf[i] * inv;
        p[i] = a;                                     // f32 attention output
        attnT[(long long)i * S + row] = (_Float16)a;  // f16 transposed copy
    }
}

// ---------------------------------------------------------------------------
// Launcher
// ---------------------------------------------------------------------------
extern "C" void kernel_launch(void* const* d_in, const int* in_sizes, int n_in,
                              void* d_out, int out_size, void* d_ws, size_t ws_size,
                              hipStream_t stream) {
    const int S = SEQ_LEN, D = DIM, C = NCLASSES;
    const float inv_sqrt_d = 0.03125f;  // 1/sqrt(1024)

    const float* x     = (const float*)d_in[0];
    const float* Wq    = (const float*)d_in[1];
    const float* bq    = (const float*)d_in[2];
    const float* gq    = (const float*)d_in[3];
    const float* betaq = (const float*)d_in[4];
    const float* Wk    = (const float*)d_in[5];
    const float* bk    = (const float*)d_in[6];
    const float* gk    = (const float*)d_in[7];
    const float* betak = (const float*)d_in[8];
    const float* Wv    = (const float*)d_in[9];
    const float* bv    = (const float*)d_in[10];
    const float* gv    = (const float*)d_in[11];
    const float* betav = (const float*)d_in[12];
    const float* Wo    = (const float*)d_in[13];
    const float* bo    = (const float*)d_in[14];
    const float* Wc    = (const float*)d_in[15];
    const float* bc    = (const float*)d_in[16];

    float* o_out    = (float*)d_out;                      // [S, D]
    float* c_out    = o_out + (long long)S * D;           // [S, C]
    float* attn_out = c_out + (long long)S * C;           // [S, S]

    // ---- workspace carve-up ----
    char* w = (char*)d_ws;
    _Float16* x16   = (_Float16*)w;  w += (long long)S * D * 2;   // 16 MB
    _Float16* WqT   = (_Float16*)w;  w += (long long)D * D * 2;   //  2 MB
    _Float16* WkT   = (_Float16*)w;  w += (long long)D * D * 2;
    _Float16* WvT   = (_Float16*)w;  w += (long long)D * D * 2;
    _Float16* WoT   = (_Float16*)w;  w += (long long)D * D * 2;
    float*    tmp   = (float*)w;     w += (long long)S * D * 4;   // 32 MB
    _Float16* q16   = (_Float16*)w;  w += (long long)S * D * 2;   // 16 MB
    _Float16* k16   = (_Float16*)w;  w += (long long)S * D * 2;
    _Float16* vT16  = (_Float16*)w;  w += (long long)S * D * 2;   // [D, S]
    _Float16* val16 = (_Float16*)w;  w += (long long)S * D * 2;
    _Float16* aT16  = (_Float16*)w;  w += (long long)S * S * 2;   // 128 MB
    if ((size_t)(w - (char*)d_ws) > ws_size) return;  // deterministic guard

    // 1) precision staging: x -> f16 ; weights -> f16 transposed
    cast_f16_kernel<<<2048, 256, 0, stream>>>(x, x16, (long long)S * D);
    const int wBlocks = (D * D + 255) / 256;
    cast_transpose_kernel<<<wBlocks, 256, 0, stream>>>(Wq, WqT, D, D);
    cast_transpose_kernel<<<wBlocks, 256, 0, stream>>>(Wk, WkT, D, D);
    cast_transpose_kernel<<<wBlocks, 256, 0, stream>>>(Wv, WvT, D, D);
    cast_transpose_kernel<<<wBlocks, 256, 0, stream>>>(Wo, WoT, D, D);

    // 2) classifier head
    classifier_kernel<<<S, 256, 0, stream>>>(x, Wc, bc, c_out, D);

    // 3) q / k / v embed chains (GEMM -> f32 tmp -> LN -> f16)
    dim3 gProj(S / TILE_M, D / TILE_N);
    gemm_wmma_tdm_kernel<<<gProj, 256, 0, stream>>>(x16, WqT, tmp, nullptr, bq, 1.f,
                                                    S, D, D, D, D, D);
    ln_cast_kernel<<<S, 256, 0, stream>>>(tmp, gq, betaq, q16, D, 0, S);

    gemm_wmma_tdm_kernel<<<gProj, 256, 0, stream>>>(x16, WkT, tmp, nullptr, bk, 1.f,
                                                    S, D, D, D, D, D);
    ln_cast_kernel<<<S, 256, 0, stream>>>(tmp, gk, betak, k16, D, 0, S);

    gemm_wmma_tdm_kernel<<<gProj, 256, 0, stream>>>(x16, WvT, tmp, nullptr, bv, 1.f,
                                                    S, D, D, D, D, D);
    ln_cast_kernel<<<S, 256, 0, stream>>>(tmp, gv, betav, vT16, D, 1, S);   // v^T

    // 4) logits = q @ k^T / sqrt(d)  (k row-major == Bt form) -> attn region
    dim3 gLogit(S / TILE_M, S / TILE_N);
    gemm_wmma_tdm_kernel<<<gLogit, 256, 0, stream>>>(q16, k16, attn_out, nullptr,
                                                     nullptr, inv_sqrt_d,
                                                     S, S, D, D, D, S);

    // 5) softmax rows; write f32 attention output + f16 transposed copy
    softmax_kernel<<<S, 256, 0, stream>>>(attn_out, aT16, S);

    // 6) values = attention^T @ v  ==  aT16[S,S] @ (vT16[D,S])^T  -> f16
    dim3 gVal(S / TILE_M, D / TILE_N);
    gemm_wmma_tdm_kernel<<<gVal, 256, 0, stream>>>(aT16, vT16, nullptr, val16,
                                                   nullptr, 1.f,
                                                   S, D, S, S, S, D);

    // 7) o = values @ Wo + bo
    gemm_wmma_tdm_kernel<<<gVal, 256, 0, stream>>>(val16, WoT, o_out, nullptr, bo, 1.f,
                                                   S, D, D, D, D, D);
}
